// WSDNetDense_66520453480791
// MI455X (gfx1250) — matrix operand order repaired and checked by reading, not compiled, not run on hardware
//
#include <hip/hip_runtime.h>

// ---------------------------------------------------------------------------
// WSDNet dense head for MI455X (gfx1250, wave32, WMMA).
// Memory-bound (~195MB HBM streamed, ~12.3 GFLOP): stream W2/W_slm once as
// fp32, convert to bf16 in registers, WMMA bf16 GEMMs. vT stored bf16 and
// L2-resident for the coalesced gather. Software-pipelined staging with
// double-buffered LDS *and* double-buffered A registers so the WMMA block
// never waits on the in-flight weight stream.
// ---------------------------------------------------------------------------

typedef __attribute__((ext_vector_type(16))) __bf16 v16bf;
typedef __attribute__((ext_vector_type(8)))  __bf16 v8bf;
typedef __attribute__((ext_vector_type(4)))  __bf16 v4bf;
typedef __attribute__((ext_vector_type(8)))  float  v8f;

#define Bsz 2
#define Lsz 64
#define Dsz 1024
#define Hsz 512
#define Ssz 50000
#define Vsz 40000
#define Ksz 32
#define Mrows 128            // B*L
#define SLM_SCALE 0.1f
#define BN_EPS 1e-5f

#define SLOT_STRIDE 24       // bf16 elems per lane-slot (16 data + 8 pad = 48B)

// ---------------------------------------------------------------------------
// Kernel 1: BatchNorm statistics. One block per L channel; reduce over (B,D).
// ---------------------------------------------------------------------------
__global__ __launch_bounds__(256)
void bn_stats_kernel(const float* __restrict__ x, float* __restrict__ mean,
                     float* __restrict__ rstd) {
  const int l = blockIdx.x;
  const int tid = threadIdx.x;
  __shared__ float s1[256];
  __shared__ float s2[256];
  float s = 0.f, q = 0.f;
  for (int i = tid; i < Bsz * Dsz; i += 256) {
    const int b = i >> 10;
    const int d = i & (Dsz - 1);
    const float xv = x[((size_t)b * Lsz + l) * Dsz + d];
    s += xv;
    q += xv * xv;
  }
  s1[tid] = s; s2[tid] = q;
  __syncthreads();
  for (int off = 128; off > 0; off >>= 1) {
    if (tid < off) { s1[tid] += s1[tid + off]; s2[tid] += s2[tid + off]; }
    __syncthreads();
  }
  if (tid == 0) {
    const float inv_n = 1.0f / (float)(Bsz * Dsz);
    const float m = s1[0] * inv_n;
    const float var = s2[0] * inv_n - m * m;
    mean[l] = m;
    rstd[l] = rsqrtf(var + BN_EPS);
  }
}

// ---------------------------------------------------------------------------
// Kernel 2: apply BN affine, emit bf16 A-matrix xbf[128,1024] (row-major).
// ---------------------------------------------------------------------------
__global__ __launch_bounds__(256)
void bn_apply_kernel(const float* __restrict__ x, const float* __restrict__ mean,
                     const float* __restrict__ rstd, const float* __restrict__ gamma,
                     const float* __restrict__ beta, __bf16* __restrict__ xbf) {
  const int i = blockIdx.x * 256 + threadIdx.x;   // quads; 128 blocks cover 131072
  const int base = i * 4;
  const int row = base >> 10;                      // D = 1024
  const int l = row & (Lsz - 1);
  const float m = mean[l], r = rstd[l], g = gamma[l], bt = beta[l];
  const float4 xv = *(const float4*)(x + base);
  v4bf o;
  o[0] = (__bf16)((xv.x - m) * r * g + bt);
  o[1] = (__bf16)((xv.y - m) * r * g + bt);
  o[2] = (__bf16)((xv.z - m) * r * g + bt);
  o[3] = (__bf16)((xv.w - m) * r * g + bt);
  *(v4bf*)(xbf + base) = o;
}

// ---------------------------------------------------------------------------
// B-tile staging, split in two phases so the 16 global loads stay in flight
// underneath the WMMA block:
//   load_cols : thread (half = tid>>7, c = tid&127) issues 16 global_load_b32
//               down one K-column (coalesced 512B across the 128 col-threads;
//               23*N*4 < 8MB so all fold into one vaddr + imm offsets).
//   cvt_store : after the WMMAs, convert to bf16 and write the 32B lane-slot
//               with two ds_store_b128.
// Slot layout matches the WMMA B fragment: slot = j*32 + half*16 + (c%16).
// ---------------------------------------------------------------------------
template <bool TAIL>
__device__ __forceinline__
void load_cols(float* __restrict__ vals, const float* __restrict__ W,
               int K, int N, int n0, int kk, int tid) {
  const int half = tid >> 7;
  const int c    = tid & 127;
  const int col  = n0 + c;
  if (!TAIL || col < N) {
    const float* wp = W + (size_t)(kk + half * 8) * N + col;
#pragma unroll
    for (int r = 0; r < 16; ++r) vals[r] = wp[(size_t)(r + (r & 8)) * N];
    if (kk + 64 < K) __builtin_prefetch(wp + (size_t)64 * N, 0, 1);
  } else {
#pragma unroll
    for (int r = 0; r < 16; ++r) vals[r] = 0.f;
  }
}

__device__ __forceinline__
void cvt_store(__bf16* __restrict__ buf, const float* __restrict__ vals, int tid) {
  const int half = tid >> 7;
  const int c    = tid & 127;
  v8bf lo, hi;
#pragma unroll
  for (int r = 0; r < 8; ++r) lo[r] = (__bf16)vals[r];
#pragma unroll
  for (int r = 0; r < 8; ++r) hi[r] = (__bf16)vals[8 + r];
  const int slot = (c >> 4) * 32 + half * 16 + (c & 15);
  __bf16* dst = buf + slot * SLOT_STRIDE;
  *(v8bf*)dst = lo;
  *(v8bf*)(dst + 8) = hi;
}

// ---------------------------------------------------------------------------
// Kernel 3 (templated): C[128,N] = act(A[128,K](bf16) * W[K,N](f32->bf16) + b)
// Block = 256 threads = 8 waves; tile 128 M x 128 N; K-steps of 32;
// double-buffered LDS + double-buffered A regs, fully software-pipelined:
//   iter kk: [A(kk+32) loads] [W(kk+32) loads] [8 WMMA on A(kk), LDS(kk)]
//            [cvt+ds_store W(kk+32)] [barrier]
// A(kk) was loaded one iteration earlier (oldest in the LOADcnt FIFO), so the
// WMMAs issue under a partial loadcnt wait while the 16 weight loads fly.
// MODE 0: +bias, ReLU, bf16 row-major (h).
// MODE 1: +bias, f32 row-major (y -> d_out).
// MODE 2: +bias, bf16 transposed (vT[N,128]); one packed b128 store per frag.
// ---------------------------------------------------------------------------
template <int MODE, bool TAIL>
__global__ __launch_bounds__(256)
void gemm_bf16_kernel(const __bf16* __restrict__ A, const float* __restrict__ W,
                      const float* __restrict__ bias, float* __restrict__ Cf,
                      __bf16* __restrict__ Cbf, int K, int N, int n0base) {
  __shared__ __bf16 ldsB[2][256 * SLOT_STRIDE];   // 2 x 12KB

  const int tid  = threadIdx.x;
  const int wave = tid >> 5;
  const int lane = tid & 31;
  const int n0   = n0base + blockIdx.x * 128;

  const v8f vzero = {0.f, 0.f, 0.f, 0.f, 0.f, 0.f, 0.f, 0.f};
  v8f acc[8];
#pragma unroll
  for (int j = 0; j < 8; ++j) acc[j] = vzero;

  // A fragment addressing (ISA 16-bit A 16x32 layout):
  // lanes 0-15: row M=lane,    K = {0..7, 16..23}
  // lanes16-31: row M=lane-16, K = {8..15, 24..31}
  const int mrow  = wave * 16 + (lane & 15);
  const int ahalf = (lane >> 4) * 8;
  const __bf16* arow = A + (size_t)mrow * K + ahalf;

  union AB { v16bf v; v8bf h[2]; };
  AB a_cur, a_nxt;

  // Prologue: stage tile 0 and preload A fragment 0.
  {
    float v0[16];
    load_cols<TAIL>(v0, W, K, N, n0, 0, tid);
    cvt_store(ldsB[0], v0, tid);
    a_cur.h[0] = *(const v8bf*)(arow);
    a_cur.h[1] = *(const v8bf*)(arow + 16);
  }
  __syncthreads();

  for (int kk = 0; kk < K; kk += 32) {
    const int cur = (kk >> 5) & 1;
    const bool have_next = (kk + 32 < K);

    // Next iteration's A fragment first (oldest in the LOADcnt FIFO), then
    // the next weight tile's 16 loads; all stay in flight during the WMMAs.
    if (have_next) {
      const __bf16* ap = arow + (kk + 32);
      a_nxt.h[0] = *(const v8bf*)(ap);
      a_nxt.h[1] = *(const v8bf*)(ap + 16);
    }
    float nvals[16];
    if (have_next) load_cols<TAIL>(nvals, W, K, N, n0, kk + 32, tid);

#pragma unroll
    for (int j = 0; j < 8; ++j) {
      AB b;
      const __bf16* bp = &ldsB[cur][(j * 32 + lane) * SLOT_STRIDE];
      b.h[0] = *(const v8bf*)(bp);
      b.h[1] = *(const v8bf*)(bp + 8);
      acc[j] = __builtin_amdgcn_wmma_f32_16x16x32_bf16(
          false, a_cur.v, false, b.v, (short)0, acc[j], false, false);
    }

    // Drain the staged weight loads only after the math has issued.
    if (have_next) {
      cvt_store(ldsB[cur ^ 1], nvals, tid);
      a_cur = a_nxt;
    }
    __syncthreads();
  }

  // C layout: VGPR i -> (M = 16w + 8*(lane>=16) + i, Ncol = lane%16)
  const int rbase = wave * 16 + ((lane >> 4) << 3);
  const int coll  = lane & 15;
#pragma unroll
  for (int j = 0; j < 8; ++j) {
    const int col = n0 + j * 16 + coll;
    if (!TAIL || col < N) {
      const float bv = bias[col];
      if (MODE == 2) {
        // 8 rows per lane are contiguous in vT[col*128 + row]: one b128 store.
        v8bf o;
#pragma unroll
        for (int i = 0; i < 8; ++i) o[i] = (__bf16)(acc[j][i] + bv);
        *(v8bf*)(Cbf + (size_t)col * Mrows + rbase) = o;
      } else {
#pragma unroll
        for (int i = 0; i < 8; ++i) {
          const int row = rbase + i;
          float val = acc[j][i] + bv;
          if (MODE == 0) {
            val = fmaxf(val, 0.f);
            Cbf[(size_t)row * N + col] = (__bf16)val;
          } else {
            Cf[(size_t)row * N + col] = val;
          }
        }
      }
    }
  }
}

// ---------------------------------------------------------------------------
// Kernel 4: out[bl,s] += 0.1 * sum_k sl_w[s,k] * vT[sl_idx[s,k]*128 + bl]
// Block = 128 threads (one per bl row) x 64 senses. idx/weights staged into
// LDS with GLOBAL_LOAD_ASYNC_TO_LDS_B128 (ASYNCcnt path); guarded scalar
// fallback for the single tail block. vT is bf16 (halves L2 gather traffic);
// every read is a coalesced 64B-per-wave L2 hit.
// ---------------------------------------------------------------------------
__global__ __launch_bounds__(128)
void gather_kernel(const __bf16* __restrict__ vT, const int* __restrict__ sl_idx,
                   const float* __restrict__ sl_w, float* __restrict__ out) {
  const int s0 = blockIdx.x * 64;
  const int bl = threadIdx.x;          // 0..127
  __shared__ int   sIdx[64 * Ksz];
  __shared__ float sW[64 * Ksz];

  if (s0 + 64 <= Ssz) {
    // Fast path: async DMA of the 2 x 8KB tiles straight into LDS.
#pragma unroll
    for (int p = 0; p < 4; ++p) {
      const int t4 = (p * 128 + bl) * 4;                 // int index, 16B aligned
      const unsigned la = (unsigned)(uintptr_t)(&sIdx[t4]);
      const unsigned lw = (unsigned)(uintptr_t)(&sW[t4]);
      const unsigned long long ga =
          (unsigned long long)(uintptr_t)(sl_idx + (size_t)s0 * Ksz + t4);
      const unsigned long long gw =
          (unsigned long long)(uintptr_t)(sl_w + (size_t)s0 * Ksz + t4);
      asm volatile("global_load_async_to_lds_b128 %0, %1, off"
                   :: "v"(la), "v"(ga) : "memory");
      asm volatile("global_load_async_to_lds_b128 %0, %1, off"
                   :: "v"(lw), "v"(gw) : "memory");
    }
    asm volatile("s_wait_asynccnt 0x0" ::: "memory");
  } else {
    for (int t = bl; t < 64 * Ksz; t += 128) {
      const int s = s0 + (t >> 5);
      const bool ok = (s < Ssz);
      sIdx[t] = ok ? sl_idx[(size_t)s * Ksz + (t & 31)] : 0;
      sW[t]   = ok ? sl_w[(size_t)s * Ksz + (t & 31)] : 0.f;
    }
  }
  __syncthreads();

  for (int si = 0; si < 64; ++si) {
    const int s = s0 + si;
    if (s >= Ssz) break;
    float acc = 0.f;
#pragma unroll 8
    for (int k = 0; k < Ksz; ++k) {
      acc += sW[si * Ksz + k] *
             (float)vT[(size_t)sIdx[si * Ksz + k] * Mrows + bl];
    }
    out[(size_t)bl * Ssz + s] += SLM_SCALE * acc;
  }
}

// ---------------------------------------------------------------------------
// Host launch
// ---------------------------------------------------------------------------
extern "C" void kernel_launch(void* const* d_in, const int* in_sizes, int n_in,
                              void* d_out, int out_size, void* d_ws, size_t ws_size,
                              hipStream_t stream) {
  const float* x     = (const float*)d_in[0];
  const float* gamma = (const float*)d_in[1];
  const float* beta  = (const float*)d_in[2];
  const float* W1    = (const float*)d_in[3];
  const float* b1    = (const float*)d_in[4];
  const float* W2    = (const float*)d_in[5];
  const float* b2    = (const float*)d_in[6];
  const float* Wslm  = (const float*)d_in[7];
  const float* bslm  = (const float*)d_in[8];
  const float* slw   = (const float*)d_in[9];
  const int*   sidx  = (const int*)d_in[10];
  float* out = (float*)d_out;

  // Workspace layout (all regions fully written before being read):
  char* ws = (char*)d_ws;
  float*  mean = (float*)ws;                                   // 64 f32
  float*  rstd = mean + 64;                                    // 64 f32
  __bf16* xbf  = (__bf16*)(ws + 1024);                         // 128x1024 bf16
  __bf16* hbf  = (__bf16*)(ws + 1024 + 262144);                // 128x512  bf16
  __bf16* vT   = (__bf16*)(ws + 1024 + 262144 + 131072);       // 40064x128 bf16

  // 1) BN stats (per L channel over B*D)
  bn_stats_kernel<<<Lsz, 256, 0, stream>>>(x, mean, rstd);
  // 2) normalize -> bf16 A matrix
  bn_apply_kernel<<<(Mrows * Dsz) / (256 * 4), 256, 0, stream>>>(x, mean, rstd,
                                                                 gamma, beta, xbf);
  // 3) h = relu(xn*W1 + b1)    M=128, K=1024, N=512  (4 full tiles)
  gemm_bf16_kernel<0, false><<<Hsz / 128, 256, 0, stream>>>(xbf, W1, b1, nullptr,
                                                            hbf, Dsz, Hsz, 0);
  // 4) vT = bf16((h*W_slm + b_slm)^T)  M=128, K=512, N=40000 (312 full + tail)
  gemm_bf16_kernel<2, false><<<Vsz / 128, 256, 0, stream>>>(hbf, Wslm, bslm,
                                                            nullptr, vT, Hsz, Vsz, 0);
  gemm_bf16_kernel<2, true><<<1, 256, 0, stream>>>(hbf, Wslm, bslm, nullptr, vT,
                                                   Hsz, Vsz, (Vsz / 128) * 128);
  // 5) out = h*W2 + b2          M=128, K=512, N=50000 (390 full + tail)
  gemm_bf16_kernel<1, false><<<Ssz / 128, 256, 0, stream>>>(hbf, W2, b2, out,
                                                            nullptr, Hsz, Ssz, 0);
  gemm_bf16_kernel<1, true><<<1, 256, 0, stream>>>(hbf, W2, b2, out, nullptr,
                                                   Hsz, Ssz, (Ssz / 128) * 128);
  // 6) out += 0.1 * weighted gather over bf16 vT (L2 resident)
  gather_kernel<<<(Ssz + 63) / 64, 128, 0, stream>>>(vT, sidx, slw, out);
}
